// HeteroGAT_47974784696658
// MI455X (gfx1250) — matrix operand order repaired and checked by reading, compiled, not run on hardware
//
#include <hip/hip_runtime.h>

// ---------------------------------------------------------------------------
// Problem constants (match reference setup_inputs)
// ---------------------------------------------------------------------------
#define Nn      100000
#define Bg      1024
#define Tt      4
#define Ee      800000
#define NEG_SLOPE 0.2f
// N padded to a multiple of 128 (rows per GEMM block)
#define Np      100096   // 782 * 128

typedef __bf16 bf16;
typedef bf16  v16bf __attribute__((ext_vector_type(16)));
typedef bf16  v8bf  __attribute__((ext_vector_type(8)));
typedef float v8f   __attribute__((ext_vector_type(8)));

static_assert(sizeof(bf16) == 2, "bf16 must be 2 bytes");

__device__ __forceinline__ bf16 f2bf(float f) {
    union { float f; unsigned u; } a; a.f = f;
    unsigned r = (a.u + 0x7FFFu + ((a.u >> 16) & 1u)) >> 16;   // RNE
    unsigned short s = (unsigned short)r;
    return __builtin_bit_cast(bf16, s);
}

// monotone float <-> unsigned mapping for atomic max on floats
__device__ __forceinline__ unsigned ordf(float f) {
    unsigned u = __float_as_uint(f);
    return (u & 0x80000000u) ? ~u : (u | 0x80000000u);
}
__device__ __forceinline__ float deordf(unsigned u) {
    unsigned b = (u & 0x80000000u) ? (u & 0x7FFFFFFFu) : ~u;
    return __uint_as_float(b);
}

// ---------------------------------------------------------------------------
// 0) gather node embeddings -> bf16 h (stride 32)
// ---------------------------------------------------------------------------
__global__ void k_gather(const int* __restrict__ node_ids,
                         const float* __restrict__ table,
                         bf16* __restrict__ hbf) {
    int i = blockIdx.x * blockDim.x + threadIdx.x;
    if (i >= Nn * 32) return;
    int node = i >> 5, j = i & 31;
    hbf[i] = f2bf(table[(size_t)node_ids[node] * 32 + j]);
}

// ---------------------------------------------------------------------------
// 1) feat = h(bf16)[Np x K] @ W(f32->bf16)[K x 64]  via WMMA bf16
//    one wave computes a 16x64 strip; 8 waves -> 128 rows per block
// ---------------------------------------------------------------------------
__global__ __launch_bounds__(256)
void k_gemm_feat(const bf16* __restrict__ A, const float* __restrict__ Wg,
                 float* __restrict__ feat, int K) {
    __shared__ bf16 Bt[64 * 72];                 // W transposed: [col][K], pad 8
    int tid = threadIdx.x;
    for (int i = tid; i < K * 64; i += 256) {
        int k = i >> 6, n = i & 63;
        Bt[n * 72 + k] = f2bf(Wg[i]);
    }
    __syncthreads();

    int wave  = tid >> 5;
    int lane  = tid & 31;
    int l16   = lane & 15;
    int khalf = lane >> 4;                        // 0 or 1
    int rowBase = blockIdx.x * 128 + wave * 16;
    int arow    = rowBase + l16;                  // A-fragment row for this lane

    union AF { v16bf v; v8bf h[2]; };
    v8f acc[4] = {};

    for (int ks = 0; ks < K; ks += 32) {
        AF af;
        const bf16* ap = A + (size_t)arow * K + ks + khalf * 8;
        af.h[0] = *(const v8bf*)(ap);             // K = ks + khalf*8 + [0..7]
        af.h[1] = *(const v8bf*)(ap + 16);        // K = ks + 16 + khalf*8 + [0..7]
#pragma unroll
        for (int ct = 0; ct < 4; ++ct) {
            int col = ct * 16 + l16;
            AF bfr;
            const bf16* bp = &Bt[col * 72 + ks + khalf * 8];
            bfr.h[0] = *(const v8bf*)(bp);        // K = ks + khalf*8 + [0..7]
            bfr.h[1] = *(const v8bf*)(bp + 16);   // K = ks + 16 + khalf*8 + [0..7]
            acc[ct] = __builtin_amdgcn_wmma_f32_16x16x32_bf16(
                false, af.v, false, bfr.v, (short)0, acc[ct], false, false);
        }
    }
#pragma unroll
    for (int ct = 0; ct < 4; ++ct) {
        int col = ct * 16 + l16;
#pragma unroll
        for (int r = 0; r < 8; ++r) {
            int row = rowBase + khalf * 8 + r;    // C layout: lanes 0-15 M=r, 16-31 M=r+8
            feat[(size_t)row * 64 + col] = acc[ct][r];
        }
    }
}

// ---------------------------------------------------------------------------
// 2) el[n,h] = sum_d feat[n,h,d]*al[h,d];  er likewise
// ---------------------------------------------------------------------------
__global__ void k_el_er(const float* __restrict__ feat,
                        const float* __restrict__ al, const float* __restrict__ ar,
                        float* __restrict__ el, float* __restrict__ er) {
    int i = blockIdx.x * blockDim.x + threadIdx.x;   // node*4 + h
    if (i >= Nn * 4) return;
    int node = i >> 2, h = i & 3;
    const float* f = feat + (size_t)node * 64 + h * 16;
    const float* a = al + h * 16;
    const float* b = ar + h * 16;
    float sl = 0.f, sr = 0.f;
#pragma unroll
    for (int d = 0; d < 16; ++d) { float v = f[d]; sl += v * a[d]; sr += v * b[d]; }
    el[i] = sl; er[i] = sr;
}

// ---------------------------------------------------------------------------
// 3a) e = leakyrelu(el[s]+er[d]); store; segment-max into mord (ordered uints)
// ---------------------------------------------------------------------------
__global__ void k_edge_a(const int* __restrict__ src, const int* __restrict__ dst,
                         const float* __restrict__ el, const float* __restrict__ er,
                         float* __restrict__ ebuf, unsigned* __restrict__ mord) {
    int e = blockIdx.x * blockDim.x + threadIdx.x;
    if (e >= Ee) return;
    int s = src[e], d = dst[e];
    float4 l = *(const float4*)(el + (size_t)s * 4);
    float4 r = *(const float4*)(er + (size_t)d * 4);
    float4 v;
    v.x = l.x + r.x; v.y = l.y + r.y; v.z = l.z + r.z; v.w = l.w + r.w;
    v.x = v.x > 0.f ? v.x : NEG_SLOPE * v.x;
    v.y = v.y > 0.f ? v.y : NEG_SLOPE * v.y;
    v.z = v.z > 0.f ? v.z : NEG_SLOPE * v.z;
    v.w = v.w > 0.f ? v.w : NEG_SLOPE * v.w;
    *(float4*)(ebuf + (size_t)e * 4) = v;
    unsigned* m = mord + (size_t)d * 4;
    atomicMax(m + 0, ordf(v.x));
    atomicMax(m + 1, ordf(v.y));
    atomicMax(m + 2, ordf(v.z));
    atomicMax(m + 3, ordf(v.w));
}

// ---------------------------------------------------------------------------
// 3b) w = exp(e - m[d]); store; segment-sum into z
// ---------------------------------------------------------------------------
__global__ void k_edge_b(const int* __restrict__ dst,
                         float* __restrict__ ebuf, const unsigned* __restrict__ mord,
                         float* __restrict__ z) {
    int e = blockIdx.x * blockDim.x + threadIdx.x;
    if (e >= Ee) return;
    int d = dst[e];
    float4 v = *(const float4*)(ebuf + (size_t)e * 4);
    const unsigned* m = mord + (size_t)d * 4;
    float4 w;
    w.x = __expf(v.x - deordf(m[0]));
    w.y = __expf(v.y - deordf(m[1]));
    w.z = __expf(v.z - deordf(m[2]));
    w.w = __expf(v.w - deordf(m[3]));
    *(float4*)(ebuf + (size_t)e * 4) = w;
    float* zp = z + (size_t)d * 4;
    atomicAdd(zp + 0, w.x);
    atomicAdd(zp + 1, w.y);
    atomicAdd(zp + 2, w.z);
    atomicAdd(zp + 3, w.w);
}

// ---------------------------------------------------------------------------
// 3c) out[dst] += feat[src] * (w / z[dst]);  one wave32 per edge, 2 cols/lane
// ---------------------------------------------------------------------------
__global__ void k_edge_c(const int* __restrict__ src, const int* __restrict__ dst,
                         const float* __restrict__ ebuf, const float* __restrict__ z,
                         const float* __restrict__ feat, float* __restrict__ out) {
    int t = blockIdx.x * blockDim.x + threadIdx.x;
    int e = t >> 5;
    int lane = t & 31;
    if (e >= Ee) return;
    int s = src[e], d = dst[e];
#pragma unroll
    for (int half = 0; half < 2; ++half) {
        int col = lane + half * 32;
        int h = col >> 4;
        float alpha = ebuf[(size_t)e * 4 + h] / z[(size_t)d * 4 + h];
        atomicAdd(&out[(size_t)d * 64 + col], feat[(size_t)s * 64 + col] * alpha);
    }
}

// ---------------------------------------------------------------------------
// 4) h = (relu?)(out + sum_t b[t]); also write bf16 copy (stride 64)
// ---------------------------------------------------------------------------
__global__ void k_bias_act(float* __restrict__ h, const float* __restrict__ btab,
                           bf16* __restrict__ hbf, int relu) {
    int i = blockIdx.x * blockDim.x + threadIdx.x;
    if (i >= Nn * 64) return;
    int j = i & 63;
    float v = h[i] + btab[j] + btab[64 + j] + btab[128 + j] + btab[192 + j];
    if (relu) v = fmaxf(v, 0.f);
    h[i] = v;
    hbf[i] = f2bf(v);
}

// ---------------------------------------------------------------------------
// 5) graph pooling: embed[g] += h[n]
// ---------------------------------------------------------------------------
__global__ void k_pool(const float* __restrict__ h, const int* __restrict__ gids,
                       float* __restrict__ embed) {
    int i = blockIdx.x * blockDim.x + threadIdx.x;
    if (i >= Nn * 64) return;
    int node = i >> 6, j = i & 63;
    atomicAdd(&embed[(size_t)gids[node] * 64 + j], h[i]);
}

// ---------------------------------------------------------------------------
// 6) score = (embed @ ow1 + ob1) @ ow2 + ob2
// ---------------------------------------------------------------------------
__global__ void k_mlp(const float* __restrict__ embed,
                      const float* __restrict__ ow1, const float* __restrict__ ob1,
                      const float* __restrict__ ow2, const float* __restrict__ ob2,
                      float* __restrict__ score) {
    int b = blockIdx.x * blockDim.x + threadIdx.x;
    if (b >= Bg) return;
    const float* e = embed + (size_t)b * 64;
    float acc = ob2[0];
#pragma unroll 4
    for (int j = 0; j < 32; ++j) {
        float hj = ob1[j];
        for (int k = 0; k < 64; ++k) hj += e[k] * ow1[k * 32 + j];
        acc += hj * ow2[j];
    }
    score[b] = acc;
}

// ---------------------------------------------------------------------------
// Driver
// ---------------------------------------------------------------------------
extern "C" void kernel_launch(void* const* d_in, const int* in_sizes, int n_in,
                              void* d_out, int out_size, void* d_ws, size_t ws_size,
                              hipStream_t stream) {
    const int*   node_ids  = (const int*)  d_in[0];
    const int*   graph_ids = (const int*)  d_in[1];
    const int*   src       = (const int*)  d_in[2];
    const int*   dst       = (const int*)  d_in[3];
    const float* table     = (const float*)d_in[4];
    const float* Wl[3]  = {(const float*)d_in[5],  (const float*)d_in[9],  (const float*)d_in[13]};
    const float* ALl[3] = {(const float*)d_in[6],  (const float*)d_in[10], (const float*)d_in[14]};
    const float* ARl[3] = {(const float*)d_in[7],  (const float*)d_in[11], (const float*)d_in[15]};
    const float* Bl[3]  = {(const float*)d_in[8],  (const float*)d_in[12], (const float*)d_in[16]};
    const float* ow1 = (const float*)d_in[17];
    const float* ob1 = (const float*)d_in[18];
    const float* ow2 = (const float*)d_in[19];
    const float* ob2 = (const float*)d_in[20];

    // workspace carve-up (all offsets multiples of 256B)
    char* ws = (char*)d_ws;
    bf16*     hbf    = (bf16*)ws;     ws += (size_t)Np * 64 * sizeof(bf16);     // 12.8 MB
    float*    h_acc  = (float*)ws;    ws += (size_t)Np * 64 * sizeof(float);    // 25.6 MB
    float*    feat   = (float*)ws;    ws += (size_t)Np * 64 * sizeof(float);    // 25.6 MB
    float*    el     = (float*)ws;    ws += (size_t)Np * 4  * sizeof(float);
    float*    er     = (float*)ws;    ws += (size_t)Np * 4  * sizeof(float);
    unsigned* mord   = (unsigned*)ws; ws += (size_t)Np * 4  * sizeof(unsigned);
    float*    zbuf   = (float*)ws;    ws += (size_t)Np * 4  * sizeof(float);
    float*    ebuf   = (float*)ws;    ws += (size_t)Ee * 4  * sizeof(float);    // 12.8 MB

    float* embed = (float*)d_out;            // [Bg, 64]
    float* score = embed + (size_t)Bg * 64;  // [Bg]

    k_gather<<<(Nn * 32 + 255) / 256, 256, 0, stream>>>(node_ids, table, hbf);

    int K = 32;
    for (int l = 0; l < 3; ++l) {
        hipMemsetAsync(h_acc, 0, (size_t)Np * 64 * sizeof(float), stream);
        for (int t = 0; t < Tt; ++t) {
            k_gemm_feat<<<Np / 128, 256, 0, stream>>>(hbf, Wl[l] + (size_t)t * K * 64, feat, K);
            k_el_er<<<(Nn * 4 + 255) / 256, 256, 0, stream>>>(feat, ALl[l] + t * 64, ARl[l] + t * 64, el, er);
            hipMemsetAsync(mord, 0, (size_t)Np * 4 * sizeof(unsigned), stream);
            hipMemsetAsync(zbuf, 0, (size_t)Np * 4 * sizeof(float), stream);
            const int* st = src + (size_t)t * Ee;
            const int* dt = dst + (size_t)t * Ee;
            k_edge_a<<<(Ee + 255) / 256, 256, 0, stream>>>(st, dt, el, er, ebuf, mord);
            k_edge_b<<<(Ee + 255) / 256, 256, 0, stream>>>(dt, ebuf, mord, zbuf);
            k_edge_c<<<((size_t)Ee * 32 + 255) / 256, 256, 0, stream>>>(st, dt, ebuf, zbuf, feat, h_acc);
        }
        k_bias_act<<<(Nn * 64 + 255) / 256, 256, 0, stream>>>(h_acc, Bl[l], hbf, l < 2 ? 1 : 0);
        K = 64;
    }

    hipMemsetAsync(embed, 0, (size_t)Bg * 64 * sizeof(float), stream);
    k_pool<<<(Nn * 64 + 255) / 256, 256, 0, stream>>>(h_acc, graph_ids, embed);
    k_mlp<<<(Bg + 255) / 256, 256, 0, stream>>>(embed, ow1, ob1, ow2, ob2, score);
}
